// NVRecEncoder_12275016532457
// MI455X (gfx1250) — matrix operand order repaired and checked by reading, not compiled, not run on hardware
//
#include <hip/hip_runtime.h>
#include <hip/hip_fp16.h>

typedef __attribute__((ext_vector_type(16))) _Float16 v16h;
typedef __attribute__((ext_vector_type(8)))  float    v8f;

#define B_SZ 256
#define S    200
#define D    256
#define M_F  32
#define KB   51      // K+1
#define PAD  249
#define SPK  224     // mixing K padded to multiple of 32
#define SPM  256     // Gr rows padded
#define H4   1024
#define NROWS (B_SZ * S)   // 51200

// ---- CDNA5 async global->LDS path (probe via __has_builtin; sync fallback) ----
#if __has_builtin(__builtin_amdgcn_global_load_async_to_lds_b128)
#define HAS_ASYNC_LDS 1
typedef __attribute__((__vector_size__(4 * sizeof(int)))) int async_v4i;
#define GAS1(p) ((__attribute__((address_space(1))) async_v4i*)(p))
#define LAS3(p) ((__attribute__((address_space(3))) async_v4i*)(p))
#else
#define HAS_ASYNC_LDS 0
#endif

#if __has_builtin(__builtin_amdgcn_s_wait_asynccnt)
#define WAIT_ASYNC() __builtin_amdgcn_s_wait_asynccnt(0)
#else
#define WAIT_ASYNC() asm volatile("s_wait_asynccnt 0" ::: "memory")
#endif

// ---------------- small prep kernels ----------------

__global__ void f32_to_f16_kernel(const float* __restrict__ src,
                                  _Float16* __restrict__ dst, int n) {
    int i = blockIdx.x * blockDim.x + threadIdx.x;
    if (i < n) dst[i] = (_Float16)src[i];
}

// Bn[m][k] = Bc[m][k] / sqrt(sum_k |Bc[m][k]|^2)
__global__ void compute_Bn_kernel(const float* __restrict__ basis_l,
                                  float2* __restrict__ Bn) {
    __shared__ float inv[M_F];
    int t = threadIdx.x;
    if (t < M_F) {
        float s = 0.f;
        for (int k = 0; k < KB; ++k) {
            float re = basis_l[(t * KB + k) * 2 + 0];
            float im = basis_l[(t * KB + k) * 2 + 1];
            s += re * re + im * im;
        }
        inv[t] = rsqrtf(s);
    }
    __syncthreads();
    for (int idx = t; idx < M_F * KB; idx += blockDim.x) {
        int m = idx / KB;
        float re = basis_l[idx * 2 + 0], im = basis_l[idx * 2 + 1];
        Bn[idx] = make_float2(re * inv[m], im * inv[m]);
    }
}

// H[p][k] = sum_m c_t[p][m] * Bn[m][k]   (c real, Bn complex)
__global__ void compute_H_kernel(const float* __restrict__ ct_l,
                                 const float2* __restrict__ Bn,
                                 float2* __restrict__ H) {
    int idx = blockIdx.x * blockDim.x + threadIdx.x;
    if (idx >= PAD * KB) return;
    int p = idx / KB, k = idx % KB;
    float re = 0.f, im = 0.f;
    for (int m = 0; m < M_F; ++m) {
        float c = ct_l[p * M_F + m];
        float2 bn = Bn[m * KB + k];
        re += c * bn.x;
        im += c * bn.y;
    }
    H[idx] = make_float2(re, im);
}

// F[p][r] = igft[p][r] * (sum_k H[p][k] * Lmat[r][k])
__global__ void compute_F_kernel(const float2* __restrict__ H,
                                 const float2* __restrict__ Lmat,
                                 const float2* __restrict__ igft,
                                 float2* __restrict__ F) {
    int idx = blockIdx.x * blockDim.x + threadIdx.x;
    if (idx >= PAD * PAD) return;
    int p = idx / PAD, r = idx % PAD;
    float tre = 0.f, tim = 0.f;
    for (int k = 0; k < KB; ++k) {
        float2 h = H[p * KB + k];
        float2 lm = Lmat[r * KB + k];
        tre += h.x * lm.x - h.y * lm.y;
        tim += h.x * lm.y + h.y * lm.x;
    }
    float2 g = igft[idx];
    F[idx] = make_float2(g.x * tre - g.y * tim, g.x * tim + g.y * tre);
}

// Gr[p][q] = Re( sum_r F[p][r] * gft[r][q] )  for p,q < S, zero-padded to SPM x SPK
__global__ void compute_Gr_kernel(const float2* __restrict__ F,
                                  const float2* __restrict__ gft,
                                  _Float16* __restrict__ Grh) {
    int idx = blockIdx.x * blockDim.x + threadIdx.x;
    if (idx >= SPM * SPK) return;
    int p = idx / SPK, q = idx % SPK;
    float s = 0.f;
    if (p < S && q < S) {
        for (int r = 0; r < PAD; ++r) {
            float2 f = F[p * PAD + r];
            float2 g = gft[r * PAD + q];
            s += f.x * g.x - f.y * g.y;
        }
    }
    Grh[idx] = (_Float16)s;
}

// masked copy: hmask f32 [B][S][D], hmh f16 [B][SPK][D] (zero-padded rows)
__global__ void mask_pack_kernel(const float* __restrict__ h,
                                 const unsigned char* __restrict__ mask,
                                 float* __restrict__ hmask,
                                 _Float16* __restrict__ hmh) {
    int idx = blockIdx.x * blockDim.x + threadIdx.x;
    if (idx >= B_SZ * SPK * D) return;
    int d = idx % D;
    int q = (idx / D) % SPK;
    int b = idx / (D * SPK);
    if (q < S) {
        float v = mask[b * S + q] ? 0.f : h[((long)b * S + q) * D + d];
        hmask[((long)b * S + q) * D + d] = v;
        hmh[idx] = (_Float16)v;
    } else {
        hmh[idx] = (_Float16)0.f;
    }
}

// out = LayerNorm(X + R) * w + b ; optional f16 copy
__global__ __launch_bounds__(256) void add_ln_kernel(
    const float* __restrict__ X, const float* __restrict__ R,
    const float* __restrict__ w, const float* __restrict__ b,
    float* __restrict__ outf, _Float16* __restrict__ outh) {
    __shared__ float red[D];
    int row = blockIdx.x, t = threadIdx.x;
    long base = (long)row * D + t;
    float v = X[base] + R[base];
    red[t] = v;
    __syncthreads();
    for (int s = D / 2; s > 0; s >>= 1) {
        if (t < s) red[t] += red[t + s];
        __syncthreads();
    }
    float mu = red[0] * (1.0f / D);
    __syncthreads();
    float dv = v - mu;
    red[t] = dv * dv;
    __syncthreads();
    for (int s = D / 2; s > 0; s >>= 1) {
        if (t < s) red[t] += red[t + s];
        __syncthreads();
    }
    float var = red[0] * (1.0f / D);
    float y = dv * rsqrtf(var + 1e-12f) * w[t] + b[t];
    outf[base] = y;
    if (outh) outh[base] = (_Float16)y;
}

// ---------------- generic WMMA f16 GEMM, double-buffered ----------------
// C[M][N] = act( A_h[M][K] @ B_h[K][N] + bias ), f32 accumulate via v_wmma_f32_16x16x32_f16
// Block tile 64x128, 8 waves (wave = 16x64), K step 32.
// A tile: straight copy -> async global->LDS (ASYNCcnt) when available.
// B tile: global->VGPR->LDS transpose ([n][k]); overlapped with WMMA via double buffering.
#define TM 64
#define TN 128
#define TK 32

__global__ __launch_bounds__(256) void gemm_wmma_kernel(
    const _Float16* __restrict__ A, int lda, long strideA,
    const _Float16* __restrict__ Bm, int ldb, long strideB,
    float* __restrict__ Cf, _Float16* __restrict__ Ch, int ldc, long strideC,
    const float* __restrict__ bias, int Mvalid, int K, int applyGelu) {
    __shared__ _Float16 Alds[2][TM][TK];   // [buf][m][k]
    __shared__ _Float16 Blds[2][TN][TK];   // [buf][n][k] (transposed)

    int bM = blockIdx.x, bN = blockIdx.y, bz = blockIdx.z;
    const _Float16* Ab = A + (long)bz * strideA + (long)bM * TM * lda;
    const _Float16* Bb = Bm + (long)bz * strideB + (long)bN * TN;

    int t = threadIdx.x;
    int lane = t & 31;
    int wave = t >> 5;
    int wM = wave & 3;   // 4 waves along M (16 rows each)
    int wN = wave >> 2;  // 2 waves along N (64 cols each)
    int lhalf = lane & 15;
    int hi = lane >> 4;  // 0 or 1

    // per-thread staging coordinates
    int arow = t >> 2;             // 0..63
    int aseg = (t & 3) * 8;        // 0,8,16,24
    int k0r = t >> 4;              // B seg0: krow 0..15
    int ns0 = (t & 15) * 8;        // B seg0: n 0..120
    int k1r = k0r + 16;            // B seg1: krow 16..31

    v8f acc[4];
#pragma unroll
    for (int i = 0; i < 4; ++i)
#pragma unroll
        for (int j = 0; j < 8; ++j) acc[i][j] = 0.f;

    uint4 breg0, breg1;
#if !HAS_ASYNC_LDS
    uint4 areg;
#endif

    // ---- prologue: stage k0 = 0 into buffer 0 ----
    {
#if HAS_ASYNC_LDS
        __builtin_amdgcn_global_load_async_to_lds_b128(
            GAS1(Ab + (long)arow * lda + aseg),
            LAS3(&Alds[0][arow][aseg]), 0, 0);
#else
        areg = *(const uint4*)(Ab + (long)arow * lda + aseg);
#endif
        breg0 = *(const uint4*)(Bb + (long)k0r * ldb + ns0);
        breg1 = *(const uint4*)(Bb + (long)k1r * ldb + ns0);
#if !HAS_ASYNC_LDS
        *(uint4*)(&Alds[0][arow][aseg]) = areg;
#endif
        _Float16 tmp[8];
        *(uint4*)tmp = breg0;
#pragma unroll
        for (int j = 0; j < 8; ++j) Blds[0][ns0 + j][k0r] = tmp[j];
        *(uint4*)tmp = breg1;
#pragma unroll
        for (int j = 0; j < 8; ++j) Blds[0][ns0 + j][k1r] = tmp[j];
#if HAS_ASYNC_LDS
        WAIT_ASYNC();
#endif
        __syncthreads();
    }

    int cur = 0;
    for (int k0 = 0; k0 < K; k0 += TK) {
        int nxt = cur ^ 1;
        bool hasNext = (k0 + TK) < K;
        if (hasNext) {
            int kn = k0 + TK;
#if HAS_ASYNC_LDS
            __builtin_amdgcn_global_load_async_to_lds_b128(
                GAS1(Ab + (long)arow * lda + kn + aseg),
                LAS3(&Alds[nxt][arow][aseg]), 0, 0);
#else
            areg = *(const uint4*)(Ab + (long)arow * lda + kn + aseg);
#endif
            breg0 = *(const uint4*)(Bb + (long)(kn + k0r) * ldb + ns0);
            breg1 = *(const uint4*)(Bb + (long)(kn + k1r) * ldb + ns0);
            if (kn + TK < K)
                __builtin_prefetch(Bb + (long)(kn + TK + k0r) * ldb + ns0, 0, 1);
        }

        // ---- compute from buffer `cur` ----
        // A fragment: lanes 0-15 -> K {0..15}; lanes 16-31 -> K {16..31}
        v16h af;
        {
            const _Float16* ap = &Alds[cur][wM * 16 + lhalf][0];
            int kb = hi * 16;
#pragma unroll
            for (int j = 0; j < 16; ++j) af[j] = ap[kb + j];
        }
#pragma unroll
        for (int tx = 0; tx < 4; ++tx) {
            int n = wN * 64 + tx * 16 + lhalf;
            const _Float16* bp = &Blds[cur][n][hi * 16];
            v16h bf;
#pragma unroll
            for (int j = 0; j < 16; ++j) bf[j] = bp[j];
            acc[tx] = __builtin_amdgcn_wmma_f32_16x16x32_f16(
                false, af, false, bf, (short)0, acc[tx], false, false);
        }

        if (hasNext) {
#if !HAS_ASYNC_LDS
            *(uint4*)(&Alds[nxt][arow][aseg]) = areg;
#endif
            _Float16 tmp[8];
            *(uint4*)tmp = breg0;
#pragma unroll
            for (int j = 0; j < 8; ++j) Blds[nxt][ns0 + j][k0r] = tmp[j];
            *(uint4*)tmp = breg1;
#pragma unroll
            for (int j = 0; j < 8; ++j) Blds[nxt][ns0 + j][k1r] = tmp[j];
#if HAS_ASYNC_LDS
            WAIT_ASYNC();
#endif
            __syncthreads();
            cur = nxt;
        }
    }

    // epilogue: C VGPR r -> row = r + 8*hi, col = lane%16
#pragma unroll
    for (int tx = 0; tx < 4; ++tx) {
        int col = bN * TN + wN * 64 + tx * 16 + lhalf;
        float bv = bias ? bias[col] : 0.f;
#pragma unroll
        for (int r = 0; r < 8; ++r) {
            int row = bM * TM + wM * 16 + hi * 8 + r;
            if (row < Mvalid) {
                float v = acc[tx][r] + bv;
                if (applyGelu) v = 0.5f * v * (1.0f + erff(v * 0.70710678118f));
                long o = (long)bz * strideC + (long)row * ldc + col;
                if (Cf) Cf[o] = v;
                if (Ch) Ch[o] = (_Float16)v;
            }
        }
    }
}

// ---------------- launch ----------------

extern "C" void kernel_launch(void* const* d_in, const int* in_sizes, int n_in,
                              void* d_out, int out_size, void* d_ws, size_t ws_size,
                              hipStream_t stream) {
    const float*         hidden = (const float*)d_in[0];
    const unsigned char* pmask  = (const unsigned char*)d_in[1];
    const float*         basis  = (const float*)d_in[2];   // [2][32][51][2]
    const float*         c_t    = (const float*)d_in[3];   // [2][249][32]
    const float*         ln1w   = (const float*)d_in[4];
    const float*         ln1b   = (const float*)d_in[5];
    const float*         W1     = (const float*)d_in[6];   // [2][256][1024]
    const float*         b1     = (const float*)d_in[7];
    const float*         W2     = (const float*)d_in[8];   // [2][1024][256]
    const float*         b2     = (const float*)d_in[9];
    const float*         ln2w   = (const float*)d_in[10];
    const float*         ln2b   = (const float*)d_in[11];
    const float2*        gft    = (const float2*)d_in[12]; // complex64 [249][249]
    const float2*        igft   = (const float2*)d_in[13];
    const float2*        Lmat   = (const float2*)d_in[14]; // complex64 [249][51]
    float* out = (float*)d_out;

    char* ws = (char*)d_ws;
    const size_t szHSD = (size_t)B_SZ * S * D * sizeof(float);        // 52.4 MB
    float*    hmask = (float*)(ws);
    float*    hcur  = (float*)(ws + szHSD);
    float*    Xbuf  = (float*)(ws + 2 * szHSD);
    size_t off = 3 * szHSD;
    _Float16* hmh   = (_Float16*)(ws + off); off += (size_t)B_SZ * SPK * D * 2;
    _Float16* acth  = (_Float16*)(ws + off); off += (size_t)NROWS * D * 2;
    _Float16* midh  = (_Float16*)(ws + off); off += (size_t)NROWS * H4 * 2;
    _Float16* Grh   = (_Float16*)(ws + off); off += (size_t)SPM * SPK * 2;
    _Float16* W1h   = (_Float16*)(ws + off); off += (size_t)2 * D * H4 * 2;
    _Float16* W2h   = (_Float16*)(ws + off); off += (size_t)2 * H4 * D * 2;
    float2*   Bn    = (float2*)(ws + off);   off += ((size_t)M_F * KB * 8 + 255) & ~255ull;
    float2*   Hc    = (float2*)(ws + off);   off += ((size_t)PAD * KB * 8 + 255) & ~255ull;
    float2*   Fc    = (float2*)(ws + off);

    // weights -> f16 (both layers)
    {
        int n = 2 * D * H4;
        f32_to_f16_kernel<<<(n + 255) / 256, 256, 0, stream>>>(W1, W1h, n);
        f32_to_f16_kernel<<<(n + 255) / 256, 256, 0, stream>>>(W2, W2h, n);
    }

    for (int l = 0; l < 2; ++l) {
        const float* src = (l == 0) ? hidden : hcur;

        // mask + pack to f16 (K padded)
        {
            int n = B_SZ * SPK * D;
            mask_pack_kernel<<<(n + 255) / 256, 256, 0, stream>>>(src, pmask, hmask, hmh);
        }
        // mixing matrix Gr
        compute_Bn_kernel<<<1, 256, 0, stream>>>(basis + (size_t)l * M_F * KB * 2, Bn);
        compute_H_kernel<<<(PAD * KB + 255) / 256, 256, 0, stream>>>(
            c_t + (size_t)l * PAD * M_F, Bn, Hc);
        compute_F_kernel<<<(PAD * PAD + 255) / 256, 256, 0, stream>>>(Hc, Lmat, igft, Fc);
        compute_Gr_kernel<<<(SPM * SPK + 255) / 256, 256, 0, stream>>>(Fc, gft, Grh);

        // X[b] = Gr @ hm[b]    (M=200 pad 256, N=256, K=224, 256 batches)
        gemm_wmma_kernel<<<dim3(SPM / TM, D / TN, B_SZ), 256, 0, stream>>>(
            Grh, SPK, 0,
            hmh, D, (long)SPK * D,
            Xbuf, nullptr, D, (long)S * D,
            nullptr, S, SPK, 0);

        // h = LN(X + hmask); also f16 copy for FFN
        add_ln_kernel<<<NROWS, D, 0, stream>>>(Xbuf, hmask, ln1w + l * D, ln1b + l * D,
                                               hcur, acth);

        // mid = gelu(h @ W1 + b1)   (M=51200, N=1024, K=256)
        gemm_wmma_kernel<<<dim3(NROWS / TM, H4 / TN, 1), 256, 0, stream>>>(
            acth, D, 0,
            W1h + (size_t)l * D * H4, H4, 0,
            nullptr, midh, H4, 0,
            b1 + (size_t)l * H4, NROWS, D, 1);

        // Y = mid @ W2 + b2        (M=51200, N=256, K=1024)
        gemm_wmma_kernel<<<dim3(NROWS / TM, D / TN, 1), 256, 0, stream>>>(
            midh, H4, 0,
            W2h + (size_t)l * H4 * D, D, 0,
            Xbuf, nullptr, D, 0,
            b2 + (size_t)l * D, NROWS, H4, 0);

        // h = LN(Y + h)
        float* dst = (l == 1) ? out : hcur;
        add_ln_kernel<<<NROWS, D, 0, stream>>>(Xbuf, hcur, ln2w + l * D, ln2b + l * D,
                                               dst, nullptr);
    }
    (void)in_sizes; (void)n_in; (void)out_size; (void)ws_size;
}
